// FastWeightGluMLPMultihead_81784767251156
// MI455X (gfx1250) — matrix operand order repaired and checked by reading, not compiled, not run on hardware
//
#include <hip/hip_runtime.h>
#include <math.h>

// ---------------------------------------------------------------------------
// FastWeightGluMLPMultihead for MI455X (gfx1250, wave32, WMMA bf16)
// One persistent workgroup per batch-head; sequential chunk scan with all
// weights / gradients / tile intermediates resident in LDS (~317KB / WGP).
// All GEMMs via v_wmma_f32_16x16x32_bf16 with register-resident A fragments,
// contiguous-only LDS B fragments, and TDM (tensor_load_to_lds) staging of lr.
// ---------------------------------------------------------------------------

#define BH      48
#define SEQ     4096
#define CDIM    64
#define CHDIM   256
#define CS      256
#define NCHUNK  (SEQ / CS)
#define RB      64
#define NRB     (CS / RB)
#define NSTEPS  5

#define NS_A  3.4445f
#define NS_B  (-4.775f)
#define NS_C  2.0315f

typedef __bf16 bf16_t;
typedef __attribute__((ext_vector_type(16))) __bf16 v16bf;
typedef __attribute__((ext_vector_type(8)))  float  v8f;

static __device__ inline v8f zero_v8f() {
  v8f c;
#pragma unroll
  for (int i = 0; i < 8; ++i) c[i] = 0.0f;
  return c;
}

static __device__ inline v8f wmma_bf16(v16bf a, v16bf b, v8f c) {
  return __builtin_amdgcn_wmma_f32_16x16x32_bf16(
      /*neg_a=*/false, a, /*neg_b=*/false, b,
      /*c_mod=*/(short)0, c, /*reuse_a=*/false, /*reuse_b=*/false);
}

// --- A fragment (16x32, row-major M x K source), ISA interleaved layout:
//     lanes 0-15: K {0..7, 16..23}; lanes 16-31: K {8..15, 24..31}
static __device__ inline v16bf frag_a_bf(const bf16_t* p, int ld, int lane) {
  const int m  = lane & 15;
  const int k0 = (lane & 16) ? 8 : 0;
  const bf16_t* r = p + m * ld + k0;
  v16bf f;
#pragma unroll
  for (int j = 0; j < 8; ++j) { f[j] = r[j]; f[8 + j] = r[16 + j]; }
  return f;
}

static __device__ inline v16bf frag_a_f32(const float* p, int ld, int lane) {
  const int m  = lane & 15;
  const int k0 = (lane & 16) ? 8 : 0;
  const float* r = p + m * ld + k0;
  v16bf f;
#pragma unroll
  for (int j = 0; j < 8; ++j) { f[j] = (bf16_t)r[j]; f[8 + j] = (bf16_t)r[16 + j]; }
  return f;
}

// --- B fragment (32x16) from transposed (N x K row-major) storage: interleaved
//     lanes 0-15: K {0..7, 16..23}; lanes 16-31: K {8..15, 24..31}; lane&15 = N column
static __device__ inline v16bf frag_b_nk(const bf16_t* p, int ld, int lane) {
  const int n  = lane & 15;
  const int kb = (lane & 16) ? 8 : 0;
  const bf16_t* r = p + n * ld + kb;
  v16bf f;
#pragma unroll
  for (int j = 0; j < 8; ++j) { f[j] = r[j]; f[8 + j] = r[16 + j]; }
  return f;
}

// --- C/D fragment helpers (reg r: lanes0-15 -> M=r, lanes16-31 -> M=r+8)
static __device__ inline void cstore_bf(bf16_t* p, int ld, int lane, v8f c) {
  const int n  = lane & 15;
  const int m0 = (lane & 16) ? 8 : 0;
#pragma unroll
  for (int r = 0; r < 8; ++r) p[(m0 + r) * ld + n] = (bf16_t)c[r];
}

// store D tile transposed: element (m,n) -> p[n*ldT + m]  (contiguous per lane)
static __device__ inline void cstore_bf_T(bf16_t* p, int ldT, int lane, v8f c) {
  const int n  = lane & 15;
  const int m0 = (lane & 16) ? 8 : 0;
  bf16_t* r = p + n * ldT + m0;
#pragma unroll
  for (int t = 0; t < 8; ++t) r[t] = (bf16_t)c[t];
}

static __device__ inline void cstore_f32(float* p, int ld, int lane, v8f c) {
  const int n  = lane & 15;
  const int m0 = (lane & 16) ? 8 : 0;
#pragma unroll
  for (int r = 0; r < 8; ++r) p[(m0 + r) * ld + n] = c[r];
}

static __device__ inline v8f cload_bf(const bf16_t* p, int ld, int lane) {
  const int n  = lane & 15;
  const int m0 = (lane & 16) ? 8 : 0;
  v8f c;
#pragma unroll
  for (int r = 0; r < 8; ++r) c[r] = (float)p[(m0 + r) * ld + n];
  return c;
}

static __device__ inline v8f cload_bf_scaled(const bf16_t* p, int ld, int lane, float s) {
  const int n  = lane & 15;
  const int m0 = (lane & 16) ? 8 : 0;
  v8f c;
#pragma unroll
  for (int r = 0; r < 8; ++r) c[r] = s * (float)p[(m0 + r) * ld + n];
  return c;
}

static __device__ inline float sigmoidf_(float x) { return 1.0f / (1.0f + __expf(-x)); }

// --- Tensor Data Mover: 1-D copy of 256 f32 global -> LDS (ISA 08, D# groups)
//     6-arg builtin form (clang-23 / therock headers)
static __device__ inline void tdm_copy_256f(const float* src, unsigned lds_off) {
  typedef unsigned int u32x4 __attribute__((ext_vector_type(4)));
  typedef int i32x8 __attribute__((ext_vector_type(8)));
  typedef int i32x4 __attribute__((ext_vector_type(4)));
  const unsigned long long ga = (unsigned long long)(uintptr_t)src;
  u32x4 g0;
  g0[0] = 1u;                                                   // count=1 (valid)
  g0[1] = lds_off;                                              // lds_addr
  g0[2] = (unsigned)ga;                                         // global_addr[31:0]
  g0[3] = (unsigned)((ga >> 32) & 0x01FFFFFFull) | (2u << 30);  // addr[56:32]|type=2
  i32x8 g1;
  g1[0] = (2 << 16);    // data_size = 4 bytes
  g1[1] = (256 << 16);  // tensor_dim0 = 256   (bits 63:48)
  g1[2] = (1 << 16);    // tensor_dim1 = 1     (bits 95:80)
  g1[3] = (256 << 16);  // tile_dim0 = 256     (bits 127:112)
  g1[4] = 1;            // tile_dim1 = 1       (bits 143:128)
  g1[5] = 256;          // tensor_dim0_stride  (bits 191:160)
  g1[6] = 0; g1[7] = 0;
  i32x4 z4; z4[0] = 0; z4[1] = 0; z4[2] = 0; z4[3] = 0;
  i32x8 z8;
#pragma unroll
  for (int i = 0; i < 8; ++i) z8[i] = 0;
  __builtin_amdgcn_tensor_load_to_lds(g0, g1, z4, z4, z8, 0);
}

__global__ void __launch_bounds__(256)
fastweight_glu_mlp_kernel(const float* __restrict__ q,  const float* __restrict__ k,
                          const float* __restrict__ v,  const float* __restrict__ w0,
                          const float* __restrict__ w1, const float* __restrict__ w2,
                          const float* __restrict__ lr0, const float* __restrict__ lr1,
                          const float* __restrict__ lr2, float* __restrict__ out)
{
  extern __shared__ char smem_raw[];
  bf16_t* sW0t = (bf16_t*)smem_raw;            // CH x C  (w0 transposed)
  bf16_t* sW1  = sW0t + CHDIM * CDIM;          // CH x C  (natural)
  bf16_t* sW2t = sW1  + CHDIM * CDIM;          // CH x C  (w2 transposed)
  bf16_t* tG   = sW2t + CHDIM * CDIM;          // grad: dgpre^T (CHxRB) | NS: Xn | out: gate_q
  bf16_t* tH   = tG   + RB * CHDIM;            // grad: hidden^T        | NS: Xt | out: hpre_q
  bf16_t* tD   = tH   + RB * CHDIM;            // grad: dhpre^T         | W1^T (CxCH) for out
  bf16_t* g0   = tD   + RB * CHDIM;            // C x CH
  bf16_t* g1t  = g0   + CDIM * CHDIM;          // C x CH (g1 pre-transposed)
  bf16_t* g2   = g1t  + CDIM * CHDIM;          // C x CH
  bf16_t* sKt  = g2   + CDIM * CHDIM;          // C x RB (K block^T) | NS: A
  bf16_t* sVt  = sKt  + CDIM * RB;             // C x RB (V block^T) | NS: Bm
  float*  slr0 = (float*)(sVt + CDIM * RB);    // CS (chunk lr cache via TDM)
  float*  slr1 = slr0 + CS;
  float*  slr2 = slr1 + CS;
  float*  n0   = slr2 + CS;                    // CH
  float*  n1   = n0 + CHDIM;                   // C
  float*  n2   = n1 + CDIM;                    // CH
  float*  red  = n2 + CHDIM;                   // 1

  const int bh   = blockIdx.x;
  const int tid  = threadIdx.x;
  const int lane = tid & 31;
  const int wave = tid >> 5;

  const float* qb  = q  + (size_t)bh * SEQ * CDIM;
  const float* kb  = k  + (size_t)bh * SEQ * CDIM;
  const float* vb  = v  + (size_t)bh * SEQ * CDIM;
  const float* w0b = w0 + (size_t)bh * CDIM * CHDIM;
  const float* w1b = w1 + (size_t)bh * CHDIM * CDIM;
  const float* w2b = w2 + (size_t)bh * CDIM * CHDIM;
  const float* l0b = lr0 + (size_t)bh * SEQ;
  const float* l1b = lr1 + (size_t)bh * SEQ;
  const float* l2b = lr2 + (size_t)bh * SEQ;

  // ---- stage weights into LDS (bf16; w0/w2 transposed), detached column norms
  for (int i = tid; i < CHDIM * CDIM; i += 256) {
    const int h = i >> 6, c = i & 63;
    sW0t[i] = (bf16_t)w0b[c * CHDIM + h];
    sW2t[i] = (bf16_t)w2b[c * CHDIM + h];
    sW1[i]  = (bf16_t)w1b[i];
  }
  for (int h = tid; h < CHDIM; h += 256) {
    float s0 = 0.f, s2 = 0.f;
    for (int c = 0; c < CDIM; ++c) {
      const float a0 = w0b[c * CHDIM + h]; s0 += a0 * a0;
      const float a2 = w2b[c * CHDIM + h]; s2 += a2 * a2;
    }
    n0[h] = sqrtf(s0);  n2[h] = sqrtf(s2);
  }
  if (tid < CDIM) {
    float s = 0.f;
    for (int h = 0; h < CHDIM; ++h) { const float a = w1b[h * CDIM + tid]; s += a * a; }
    n1[tid] = sqrtf(s);
  }
  __syncthreads();

  const int moW   = (wave >> 1) * 16;   // row-tile owned by this wave (4x16 grids)
  const int nbase = (wave & 1) * 8;     // first of 8 column-tiles owned

  for (int ch = 0; ch < NCHUNK; ++ch) {
    const int rowbase = ch * CS;

    // TDM: DMA this chunk's lr vectors into LDS (one wave issues; EXEC ignored)
    if (tid < 32) {
      tdm_copy_256f(l0b + rowbase, (unsigned)((char*)slr0 - smem_raw));
      tdm_copy_256f(l1b + rowbase, (unsigned)((char*)slr1 - smem_raw));
      tdm_copy_256f(l2b + rowbase, (unsigned)((char*)slr2 - smem_raw));
    }
    for (int i = tid; i < CDIM * CHDIM; i += 256) {
      g0[i] = (bf16_t)0.f; g1t[i] = (bf16_t)0.f; g2[i] = (bf16_t)0.f;
    }
    if (tid < 32) __builtin_amdgcn_s_wait_tensorcnt(0);
    __syncthreads();

    // ================= inner gradient over 64-row blocks =================
    for (int rb = 0; rb < NRB; ++rb) {
      const int r0 = rowbase + rb * RB;
      const float* Kb = kb + (size_t)r0 * CDIM;
      const float* Vb = vb + (size_t)r0 * CDIM;

      if (rb + 1 < NRB) {  // prefetch next row block
        __builtin_prefetch(kb + (size_t)(r0 + RB) * CDIM + tid * 16, 0, 0);
        __builtin_prefetch(vb + (size_t)(r0 + RB) * CDIM + tid * 16, 0, 0);
      }

      // stage K^T, V^T for the gradient outer products (bf16)
      for (int t = 0; t < 16; ++t) {
        const int i = tid * 16 + t;
        const int c = i >> 6, j = i & 63;
        sKt[i] = (bf16_t)kb[(size_t)(r0 + j) * CDIM + c];
        sVt[i] = (bf16_t)vb[(size_t)(r0 + j) * CDIM + c];
      }

      // gate=Kb@W0, hpre=Kb@W2, dhid=Vb@W1^T -> stored TRANSPOSED (CH x RB)
      {
        v16bf ak[2], av[2];
#pragma unroll
        for (int kk = 0; kk < 2; ++kk) {
          ak[kk] = frag_a_f32(Kb + moW * CDIM + kk * 32, CDIM, lane);
          av[kk] = frag_a_f32(Vb + moW * CDIM + kk * 32, CDIM, lane);
        }
        for (int j = 0; j < 8; ++j) {
          const int no = (nbase + j) * 16;
          v8f cg = zero_v8f(), chh = zero_v8f(), cd = zero_v8f();
#pragma unroll
          for (int kk = 0; kk < 2; ++kk) {
            const v16bf b0 = frag_b_nk(sW0t + no * CDIM + kk * 32, CDIM, lane);
            const v16bf b2 = frag_b_nk(sW2t + no * CDIM + kk * 32, CDIM, lane);
            const v16bf b1 = frag_b_nk(sW1  + no * CDIM + kk * 32, CDIM, lane);
            cg  = wmma_bf16(ak[kk], b0, cg);
            chh = wmma_bf16(ak[kk], b2, chh);
            cd  = wmma_bf16(av[kk], b1, cd);
          }
          cstore_bf_T(tG + no * RB + moW, RB, lane, cg);
          cstore_bf_T(tH + no * RB + moW, RB, lane, chh);
          cstore_bf_T(tD + no * RB + moW, RB, lane, cd);
        }
      }
      __syncthreads();

      // elementwise SwiGLU backward with lr folded in (layout h-major, l minor)
      {
        const int rl = rb * RB;
        for (int i = tid; i < RB * CHDIM; i += 256) {
          const int l = i & (RB - 1);
          const float L0 = slr0[rl + l], L1 = slr1[rl + l], L2 = slr2[rl + l];
          const float g = (float)tG[i], h = (float)tH[i], d = (float)tD[i];
          const float s  = sigmoidf_(g);
          const float sg = g * s;
          tH[i] = (bf16_t)(sg * h * L1);                              // hidden*l1
          tD[i] = (bf16_t)(d * sg * L2);                              // dhpre*l2
          tG[i] = (bf16_t)((d * h) * s * (1.f + g * (1.f - s)) * L0); // dgpre*l0
        }
      }
      __syncthreads();

      // g0 += K^T@dgpre' ; g1t += V^T@hidden' ; g2 += K^T@dhpre'
      {
        v16bf aK[2], aV[2];
#pragma unroll
        for (int kk = 0; kk < 2; ++kk) {
          aK[kk] = frag_a_bf(sKt + moW * RB + kk * 32, RB, lane);
          aV[kk] = frag_a_bf(sVt + moW * RB + kk * 32, RB, lane);
        }
        for (int j = 0; j < 8; ++j) {
          const int no = (nbase + j) * 16;
          v8f c0 = cload_bf(g0  + moW * CHDIM + no, CHDIM, lane);
          v8f c1 = cload_bf(g1t + moW * CHDIM + no, CHDIM, lane);
          v8f c2 = cload_bf(g2  + moW * CHDIM + no, CHDIM, lane);
#pragma unroll
          for (int kk = 0; kk < 2; ++kk) {
            const v16bf bg = frag_b_nk(tG + no * RB + kk * 32, RB, lane);
            const v16bf bh = frag_b_nk(tH + no * RB + kk * 32, RB, lane);
            const v16bf bd = frag_b_nk(tD + no * RB + kk * 32, RB, lane);
            c0 = wmma_bf16(aK[kk], bg, c0);
            c1 = wmma_bf16(aV[kk], bh, c1);
            c2 = wmma_bf16(aK[kk], bd, c2);
          }
          cstore_bf(g0  + moW * CHDIM + no, CHDIM, lane, c0);
          cstore_bf(g1t + moW * CHDIM + no, CHDIM, lane, c1);
          cstore_bf(g2  + moW * CHDIM + no, CHDIM, lane, c2);
        }
      }
      __syncthreads();
    }

    // ================= quintic Newton-Schulz (bf16), X is 64x256 ==========
    for (int m = 0; m < 3; ++m) {
      bf16_t* X  = (m == 0) ? g0 : (m == 1) ? g1t : g2;
      bf16_t* Xt = tH;    // CH x C mirror of X
      bf16_t* Xn = tG;    // next X (row-major)
      bf16_t* sA  = sKt;  // 64x64 (symmetric)
      bf16_t* sBm = sVt;  // 64x64

      if (tid == 0) red[0] = 0.f;
      __syncthreads();
      { float part = 0.f;
        for (int i = tid; i < CDIM * CHDIM; i += 256) { const float x = (float)X[i]; part += x * x; }
        atomicAdd(red, part); }   // ds_add_f32
      __syncthreads();
      { const float inv = 1.f / (sqrtf(red[0]) + 1e-7f);
        for (int i = tid; i < CDIM * CHDIM; i += 256) {
          const bf16_t xv = (bf16_t)((float)X[i] * inv);
          X[i] = xv;
          Xt[(i & 255) * CDIM + (i >> 8)] = xv;
        } }
      __syncthreads();

      for (int it = 0; it < NSTEPS; ++it) {
        // A = X X^T  (64x64, K=256) -- symmetric result
        {
          const int t0 = 2 * wave;
          const int mo = (t0 >> 2) * 16, no0 = (t0 & 3) * 16;
          v16bf a[8];
#pragma unroll
          for (int kk = 0; kk < 8; ++kk)
            a[kk] = frag_a_bf(X + mo * CHDIM + kk * 32, CHDIM, lane);
          for (int p = 0; p < 2; ++p) {
            const int no = no0 + p * 16;
            v8f c = zero_v8f();
#pragma unroll
            for (int kk = 0; kk < 8; ++kk) {
              const v16bf b = frag_b_nk(X + no * CHDIM + kk * 32, CHDIM, lane);
              c = wmma_bf16(a[kk], b, c);
            }
            cstore_bf(sA + mo * CDIM + no, CDIM, lane, c);
          }
        }
        __syncthreads();
        // A2 = A A   (A symmetric -> B operand contiguous from A itself)
        {
          const int t0 = 2 * wave;
          const int mo = (t0 >> 2) * 16, no0 = (t0 & 3) * 16;
          v16bf a[2];
#pragma unroll
          for (int kk = 0; kk < 2; ++kk)
            a[kk] = frag_a_bf(sA + mo * CDIM + kk * 32, CDIM, lane);
          for (int p = 0; p < 2; ++p) {
            const int no = no0 + p * 16;
            v8f c = zero_v8f();
#pragma unroll
            for (int kk = 0; kk < 2; ++kk) {
              const v16bf b = frag_b_nk(sA + no * CDIM + kk * 32, CDIM, lane);
              c = wmma_bf16(a[kk], b, c);
            }
            cstore_bf(sBm + mo * CDIM + no, CDIM, lane, c);
          }
        }
        __syncthreads();
        // Bm = NS_B*A + NS_C*A2
        for (int i = tid; i < CDIM * CDIM; i += 256)
          sBm[i] = (bf16_t)(NS_B * (float)sA[i] + NS_C * (float)sBm[i]);
        __syncthreads();
        // Xn = NS_A*X + Bm X   (B operand contiguous from Xt)
        {
          v16bf a[2];
#pragma unroll
          for (int kk = 0; kk < 2; ++kk)
            a[kk] = frag_a_bf(sBm + moW * CDIM + kk * 32, CDIM, lane);
          for (int j = 0; j < 8; ++j) {
            const int no = (nbase + j) * 16;
            v8f c = cload_bf_scaled(X + moW * CHDIM + no, CHDIM, lane, NS_A);
#pragma unroll
            for (int kk = 0; kk < 2; ++kk) {
              const v16bf b = frag_b_nk(Xt + no * CDIM + kk * 32, CDIM, lane);
              c = wmma_bf16(a[kk], b, c);
            }
            cstore_bf(Xn + moW * CHDIM + no, CHDIM, lane, c);
          }
        }
        __syncthreads();
        for (int i = tid; i < CDIM * CHDIM; i += 256) {
          const bf16_t xv = Xn[i];
          X[i] = xv;
          Xt[(i & 255) * CDIM + (i >> 8)] = xv;
        }
        __syncthreads();
      }
    }

    // ================= weight update + weight-norm renorm ==================
    for (int i = tid; i < CHDIM * CDIM; i += 256) {
      const int h = i >> 6, c = i & 63;
      sW0t[i] = (bf16_t)((float)sW0t[i] + (float)g0[c * CHDIM + h]);
      sW1[i]  = (bf16_t)((float)sW1[i]  + (float)g1t[c * CHDIM + h]);
      sW2t[i] = (bf16_t)((float)sW2t[i] + (float)g2[c * CHDIM + h]);
    }
    __syncthreads();
    for (int h = tid; h < CHDIM; h += 256) {
      float s0 = 0.f, s2 = 0.f;
      for (int c = 0; c < CDIM; ++c) {
        const float a0 = (float)sW0t[h * CDIM + c]; s0 += a0 * a0;
        const float a2 = (float)sW2t[h * CDIM + c]; s2 += a2 * a2;
      }
      const float f0 = n0[h] / (sqrtf(s0) + 1e-5f);
      const float f2 = n2[h] / (sqrtf(s2) + 1e-5f);
      for (int c = 0; c < CDIM; ++c) {
        sW0t[h * CDIM + c] = (bf16_t)((float)sW0t[h * CDIM + c] * f0);
        sW2t[h * CDIM + c] = (bf16_t)((float)sW2t[h * CDIM + c] * f2);
      }
    }
    if (tid < CDIM) {
      float s = 0.f;
      for (int h = 0; h < CHDIM; ++h) { const float a = (float)sW1[h * CDIM + tid]; s += a * a; }
      const float f = n1[tid] / (sqrtf(s) + 1e-5f);
      for (int h = 0; h < CHDIM; ++h)
        sW1[h * CDIM + tid] = (bf16_t)((float)sW1[h * CDIM + tid] * f);
    }
    __syncthreads();
    // W1^T (C x CH) into tD for the output GEMM's contiguous B operand
    for (int i = tid; i < CHDIM * CDIM; i += 256) {
      const int h = i >> 6, c = i & 63;
      tD[c * CHDIM + h] = sW1[i];
    }
    __syncthreads();

    // ================= output: (silu(Q W0) * (Q W2)) W1 ====================
    for (int rb = 0; rb < NRB; ++rb) {
      const int r0 = rowbase + rb * RB;
      const float* Qb = qb + (size_t)r0 * CDIM;
      {
        v16bf aq[2];
#pragma unroll
        for (int kk = 0; kk < 2; ++kk)
          aq[kk] = frag_a_f32(Qb + moW * CDIM + kk * 32, CDIM, lane);
        for (int j = 0; j < 8; ++j) {
          const int no = (nbase + j) * 16;
          v8f cg = zero_v8f(), chh = zero_v8f();
#pragma unroll
          for (int kk = 0; kk < 2; ++kk) {
            const v16bf b0 = frag_b_nk(sW0t + no * CDIM + kk * 32, CDIM, lane);
            const v16bf b2 = frag_b_nk(sW2t + no * CDIM + kk * 32, CDIM, lane);
            cg  = wmma_bf16(aq[kk], b0, cg);
            chh = wmma_bf16(aq[kk], b2, chh);
          }
          cstore_bf(tG + moW * CHDIM + no, CHDIM, lane, cg);  // natural layout
          cstore_bf(tH + moW * CHDIM + no, CHDIM, lane, chh);
        }
      }
      __syncthreads();
      for (int i = tid; i < RB * CHDIM; i += 256) {
        const float g = (float)tG[i], h = (float)tH[i];
        tG[i] = (bf16_t)(g * sigmoidf_(g) * h);   // hidden_q
      }
      __syncthreads();
      // out_b = hidden_q @ W1  (64x64, K=256) -> f32 global
      {
        const int t0 = 2 * wave;
        const int mo = (t0 >> 2) * 16, no0 = (t0 & 3) * 16;
        v16bf a[8];
#pragma unroll
        for (int kk = 0; kk < 8; ++kk)
          a[kk] = frag_a_bf(tG + mo * CHDIM + kk * 32, CHDIM, lane);
        for (int p = 0; p < 2; ++p) {
          const int no = no0 + p * 16;
          v8f c = zero_v8f();
#pragma unroll
          for (int kk = 0; kk < 8; ++kk) {
            const v16bf b = frag_b_nk(tD + no * CHDIM + kk * 32, CHDIM, lane);
            c = wmma_bf16(a[kk], b, c);
          }
          cstore_f32(out + ((size_t)bh * SEQ + r0 + mo) * CDIM + no, CDIM, lane, c);
        }
      }
      __syncthreads();
    }
  }

  // ================= final weights writeout (f32, original layouts) ========
  float* w0f = out + (size_t)BH * SEQ * CDIM + (size_t)bh * CDIM * CHDIM;
  float* w1f = out + (size_t)BH * SEQ * CDIM + (size_t)BH * CDIM * CHDIM + (size_t)bh * CHDIM * CDIM;
  float* w2f = out + (size_t)BH * SEQ * CDIM + 2 * (size_t)BH * CDIM * CHDIM + (size_t)bh * CDIM * CHDIM;
  for (int i = tid; i < CHDIM * CDIM; i += 256) {
    const int h = i >> 6, c = i & 63;
    w0f[c * CHDIM + h] = (float)sW0t[i];
    w2f[c * CHDIM + h] = (float)sW2t[i];
    w1f[i] = (float)sW1[i];
  }
}

extern "C" void kernel_launch(void* const* d_in, const int* in_sizes, int n_in,
                              void* d_out, int out_size, void* d_ws, size_t ws_size,
                              hipStream_t stream) {
  (void)in_sizes; (void)n_in; (void)out_size; (void)d_ws; (void)ws_size;
  const float* q   = (const float*)d_in[0];
  const float* k   = (const float*)d_in[1];
  const float* v   = (const float*)d_in[2];
  const float* w0  = (const float*)d_in[3];
  const float* w1  = (const float*)d_in[4];
  const float* w2  = (const float*)d_in[5];
  const float* lr0 = (const float*)d_in[6];
  const float* lr1 = (const float*)d_in[7];
  const float* lr2 = (const float*)d_in[8];
  float* out = (float*)d_out;

  const size_t smem =
      (size_t)(9 * CDIM * CHDIM + 2 * CDIM * RB) * sizeof(bf16_t) +
      (size_t)(3 * CS + CHDIM + CDIM + CHDIM + 4) * sizeof(float);

  fastweight_glu_mlp_kernel<<<dim3(BH), dim3(256), smem, stream>>>(
      q, k, v, w0, w1, w2, lr0, lr1, lr2, out);
}
